// AttentionBlock_29643864277426
// MI455X (gfx1250) — matrix operand order repaired and checked by reading, not compiled, not run on hardware
//
#include <hip/hip_runtime.h>
#include <hip/hip_bf16.h>

typedef __bf16 bf16_t;
typedef __bf16 v8bf  __attribute__((ext_vector_type(8)));
typedef __bf16 v16bf __attribute__((ext_vector_type(16)));
typedef float  v8f   __attribute__((ext_vector_type(8)));

#define SEQ     4096
#define DMODEL  2048
#define NHEAD   16
#define NKV     4
#define DKH     128
#define KVD     (NKV * DKH)   // 512
#define WINDOW  1024
#define SM_SCALE 0.08838834764831845f   // 1/sqrt(128)
#define MASKV   (-3.0e38f)

// ---- fragment loaders -------------------------------------------------------
// A 16x32 bf16 fragment in this code's k order (same as B): 16 contiguous K values (half-wave split at K=16)
__device__ __forceinline__ v16bf ldA(const bf16_t* p) {
  v8bf lo = *(const v8bf*)p;
  v8bf hi = *(const v8bf*)(p + 8);
  v16bf r;
#pragma unroll
  for (int i = 0; i < 8; ++i) { r[i] = lo[i]; r[i + 8] = hi[i]; }
  return r;
}
// B 32x16 bf16 fragment in this code's k order (same as A): 16 contiguous K values (half-wave split at K=16)
__device__ __forceinline__ v16bf ldB(const bf16_t* p) {
  v8bf lo = *(const v8bf*)p;
  v8bf hi = *(const v8bf*)(p + 8);
  v16bf r;
#pragma unroll
  for (int i = 0; i < 8; ++i) { r[i] = lo[i]; r[i + 8] = hi[i]; }
  return r;
}

__device__ __forceinline__ v8f wmma_bf16(v16bf a, v16bf b, v8f c) {
  return __builtin_amdgcn_wmma_f32_16x16x32_bf16(false, a, false, b, (short)0, c,
                                                 false, false);
}

// CDNA5 async global->LDS copy (ASYNCcnt-tracked), 16B per lane.
// lds_off = low 32 bits of generic shared pointer (= LDS byte address).
__device__ __forceinline__ void async_g2l_b128(unsigned lds_off, const bf16_t* g) {
  asm volatile("global_load_async_to_lds_b128 %0, %1, off"
               :: "v"(lds_off), "v"(g) : "memory");
}
__device__ __forceinline__ void wait_asynccnt_le4() {
  asm volatile("s_wait_asynccnt 0x4" ::: "memory");
}
__device__ __forceinline__ void wait_asynccnt_0() {
  asm volatile("s_wait_asynccnt 0x0" ::: "memory");
}

// ---- fp32 -> bf16 convert ---------------------------------------------------
__global__ __launch_bounds__(256) void cvt_f32_bf16(const float* __restrict__ in,
                                                    bf16_t* __restrict__ out, int n) {
  int i = blockIdx.x * 256 + threadIdx.x;
  if (i < n) out[i] = (bf16_t)in[i];
}

// ---- NT GEMM: C[M,N] = A[M,K] * B[N,K]^T (torch Linear) ---------------------
// block = 128 threads = 4 waves arranged 2(M) x 2(N); block tile 128x128;
// wave tile 64x64 (16 WMMAs / k-step). Weight tile double-buffered in LDS via
// async global->LDS loads; A fragments loaded straight from L2.
template <typename OutT>
__global__ __launch_bounds__(128) void gemm_nt(const bf16_t* __restrict__ A,
                                               const bf16_t* __restrict__ Bw,
                                               OutT* __restrict__ C,
                                               int M, int N, int K) {
  __shared__ __align__(16) bf16_t btile[2][128][40];  // [buf][n][k], padded stride
  const int tid  = threadIdx.x;
  const int lane = tid & 31;
  const int wave = tid >> 5;
  const int r = lane & 15, q = lane >> 4;
  const int bm = blockIdx.x * 128 + (wave & 1) * 64;  // wave's M base
  const int bn0 = blockIdx.y * 128;                   // block's N base
  const int wn = (wave >> 1) * 64;                    // wave's N offset in block

  v8f acc[4][4];
  v8f z = {0.f, 0.f, 0.f, 0.f, 0.f, 0.f, 0.f, 0.f};
#pragma unroll
  for (int i = 0; i < 4; ++i)
#pragma unroll
    for (int j = 0; j < 4; ++j) acc[i][j] = z;

  // stage one 128(n) x 32(k) weight slice: thread t copies row t (64B = 4x16B)
  auto stage = [&](int buf, int k0) {
    const bf16_t* g = Bw + (size_t)(bn0 + tid) * K + k0;
    unsigned l = (unsigned)(uintptr_t)&btile[buf][tid][0];
#pragma unroll
    for (int i = 0; i < 4; ++i) async_g2l_b128(l + i * 16, g + i * 8);
  };

  stage(0, 0);
  for (int k0 = 0; k0 < K; k0 += 32) {
    const int buf = (k0 >> 5) & 1;
    if (k0 + 32 < K) {
      stage(buf ^ 1, k0 + 32);   // prefetch next slice (overlaps compute)
      wait_asynccnt_le4();       // current slice done; next 4 may be in flight
    } else {
      wait_asynccnt_0();
    }
    __syncthreads();             // all waves' slices visible

    v16bf af[4], bfm[4];
#pragma unroll
    for (int i = 0; i < 4; ++i)
      af[i] = ldA(A + (size_t)(bm + i * 16 + r) * K + k0 + q * 16);
#pragma unroll
    for (int j = 0; j < 4; ++j)
      bfm[j] = ldB(&btile[buf][wn + j * 16 + r][q * 16]);
#pragma unroll
    for (int i = 0; i < 4; ++i)
#pragma unroll
      for (int j = 0; j < 4; ++j)
        acc[i][j] = wmma_bf16(af[i], bfm[j], acc[i][j]);

    __syncthreads();             // done reading buf before it is re-staged
  }

#pragma unroll
  for (int i = 0; i < 4; ++i)
#pragma unroll
    for (int j = 0; j < 4; ++j)
#pragma unroll
      for (int m = 0; m < 8; ++m) {
        int row = bm + i * 16 + m + 8 * q;
        int col = bn0 + wn + j * 16 + r;
        C[(size_t)row * N + col] = (OutT)acc[i][j][m];
      }
}

// ---- fused per-head RMSNorm + RoPE (in place, bf16) -------------------------
__global__ __launch_bounds__(256) void rmsnorm_rope(bf16_t* __restrict__ x,
                                                    const float* __restrict__ w,
                                                    int nheads) {
  int gw = (blockIdx.x * 256 + threadIdx.x) >> 5;   // one wave per (s, head)
  int lane = threadIdx.x & 31;
  int s = gw / nheads;
  int h = gw - s * nheads;
  if (s >= SEQ) return;
  bf16_t* p = x + (size_t)s * (nheads * DKH) + h * DKH + lane * 4;
  float v[4];
#pragma unroll
  for (int i = 0; i < 4; ++i) v[i] = (float)p[i];
  float ss = v[0] * v[0] + v[1] * v[1] + v[2] * v[2] + v[3] * v[3];
#pragma unroll
  for (int o = 16; o >= 1; o >>= 1) ss += __shfl_xor(ss, o, 32);
  float inv = rsqrtf(ss * (1.0f / DKH) + 1e-6f);
#pragma unroll
  for (int i = 0; i < 4; ++i) v[i] *= inv * w[lane * 4 + i];
  // RoPE, interleaved pairs; pair index = lane*2 + pi; freq = theta^(-p/64)
#pragma unroll
  for (int pi = 0; pi < 2; ++pi) {
    int pidx = lane * 2 + pi;
    float freq = __expf(-0.20503692772820544f * (float)pidx); // ln(5e5)/64
    float ang = (float)s * freq;
    float c, sn;
    __sincosf(ang, &c, &sn);
    float re = v[2 * pi], im = v[2 * pi + 1];
    v[2 * pi]     = re * c - im * sn;
    v[2 * pi + 1] = re * sn + im * c;
  }
#pragma unroll
  for (int i = 0; i < 4; ++i) p[i] = (bf16_t)v[i];
}

// ---- V transpose: vb[S][KVD] -> vt[KVD][S] ----------------------------------
__global__ __launch_bounds__(256) void vtranspose(const bf16_t* __restrict__ vb,
                                                  bf16_t* __restrict__ vt) {
  int idx = blockIdx.x * 256 + threadIdx.x;   // SEQ*KVD threads
  int s = idx & (SEQ - 1);
  int y = idx >> 12;
  vt[(size_t)y * SEQ + s] = vb[(size_t)s * KVD + y];
}

// ---- flash attention, sliding window, one wave per (head, 16-query tile) ----
__global__ __launch_bounds__(256) void flash_swa(const bf16_t* __restrict__ qb,
                                                 const bf16_t* __restrict__ kb,
                                                 const bf16_t* __restrict__ vt,
                                                 bf16_t* __restrict__ ao) {
  __shared__ __align__(16) bf16_t plds[8][16][40];  // per-wave P tile, padded
  const int lane = threadIdx.x & 31;
  const int wave = threadIdx.x >> 5;
  const int r = lane & 15, q = lane >> 4;
  const int gw = blockIdx.x * 8 + wave;             // NHEAD * SEQ/16 waves
  const int h = gw >> 8;                            // 256 query tiles / head
  const int qt = (gw & 255) * 16;
  const int kv = h >> 2;                            // repeat_interleave GQA map

  // Q A-fragments for K-dim 128 = 4 chunks of 32
  v16bf qf[4];
  const bf16_t* qp = qb + (size_t)(qt + r) * DMODEL + h * DKH + q * 16;
#pragma unroll
  for (int c = 0; c < 4; ++c) qf[c] = ldA(qp + c * 32);

  v8f acc[8];
  v8f z = {0.f, 0.f, 0.f, 0.f, 0.f, 0.f, 0.f, 0.f};
#pragma unroll
  for (int t = 0; t < 8; ++t) acc[t] = z;
  float mi[8], li[8];
#pragma unroll
  for (int m = 0; m < 8; ++m) { mi[m] = MASKV; li[m] = 0.0f; }

  int jstart = qt - (WINDOW - 1);
  if (jstart < 0) jstart = 0;
  jstart &= ~31;

  for (int j0 = jstart; j0 <= qt + 15; j0 += 32) {
    // ---- scores: S = Q * K^T (two 16-key n-tiles) ----
    int key0 = j0 + r;       if (key0 > SEQ - 1) key0 = SEQ - 1;
    int key1 = j0 + 16 + r;  if (key1 > SEQ - 1) key1 = SEQ - 1;
    const bf16_t* kp0 = kb + (size_t)key0 * KVD + kv * DKH + q * 16;
    const bf16_t* kp1 = kb + (size_t)key1 * KVD + kv * DKH + q * 16;
    v8f s0 = z, s1 = z;
#pragma unroll
    for (int c = 0; c < 4; ++c) {
      s0 = wmma_bf16(qf[c], ldB(kp0 + c * 32), s0);
      s1 = wmma_bf16(qf[c], ldB(kp1 + c * 32), s1);
    }
    // ---- online softmax (row = m + 8*q across 16-lane group) ----
#pragma unroll
    for (int m = 0; m < 8; ++m) {
      int row = qt + m + 8 * q;
      int jA = j0 + r, jB = j0 + 16 + r;
      float a = ((jA <= row) && (jA > row - WINDOW)) ? s0[m] * SM_SCALE : MASKV;
      float b = ((jB <= row) && (jB > row - WINDOW)) ? s1[m] * SM_SCALE : MASKV;
      float vmx = fmaxf(a, b);
#pragma unroll
      for (int o = 8; o >= 1; o >>= 1) vmx = fmaxf(vmx, __shfl_xor(vmx, o, 16));
      float mnew = fmaxf(mi[m], vmx);
      float corr = __expf(mi[m] - mnew);
      mi[m] = mnew;
      float p0 = __expf(a - mnew);
      float p1 = __expf(b - mnew);
      float rs = p0 + p1;
#pragma unroll
      for (int o = 8; o >= 1; o >>= 1) rs += __shfl_xor(rs, o, 16);
      li[m] = li[m] * corr + rs;
#pragma unroll
      for (int t = 0; t < 8; ++t) acc[t][m] *= corr;
      plds[wave][m + 8 * q][r]      = (bf16_t)p0;   // D-layout -> LDS
      plds[wave][m + 8 * q][16 + r] = (bf16_t)p1;
    }
    // re-load P as A-fragment (same-wave DS ops stay in order)
    v16bf pf = ldA(&plds[wave][r][q * 16]);
    // ---- O += P * V over 8 d-tiles; V^T rows are contiguous in keys ----
#pragma unroll
    for (int t = 0; t < 8; ++t) {
      const bf16_t* vp = vt + (size_t)(kv * DKH + t * 16 + r) * SEQ + j0 + q * 16;
      acc[t] = wmma_bf16(pf, ldB(vp), acc[t]);
    }
  }
#pragma unroll
  for (int m = 0; m < 8; ++m) li[m] = 1.0f / li[m];
#pragma unroll
  for (int t = 0; t < 8; ++t)
#pragma unroll
    for (int m = 0; m < 8; ++m)
      ao[(size_t)(qt + m + 8 * q) * DMODEL + h * DKH + t * 16 + r] =
          (bf16_t)(acc[t][m] * li[m]);
}

// ---- host launcher ----------------------------------------------------------
extern "C" void kernel_launch(void* const* d_in, const int* in_sizes, int n_in,
                              void* d_out, int out_size, void* d_ws, size_t ws_size,
                              hipStream_t stream) {
  (void)in_sizes; (void)n_in; (void)out_size; (void)ws_size;
  const float* xq = (const float*)d_in[0];
  const float* xk = (const float*)d_in[1];
  const float* xv = (const float*)d_in[2];
  const float* Wq = (const float*)d_in[3];
  const float* Wk = (const float*)d_in[4];
  const float* Wv = (const float*)d_in[5];
  const float* Wo = (const float*)d_in[6];
  const float* qw = (const float*)d_in[7];
  const float* kw = (const float*)d_in[8];

  bf16_t* p = (bf16_t*)d_ws;
  bf16_t* xq_b = p; p += (size_t)SEQ * DMODEL;
  bf16_t* xk_b = p; p += (size_t)SEQ * DMODEL;
  bf16_t* xv_b = p; p += (size_t)SEQ * DMODEL;
  bf16_t* Wq_b = p; p += (size_t)DMODEL * DMODEL;
  bf16_t* Wk_b = p; p += (size_t)KVD * DMODEL;
  bf16_t* Wv_b = p; p += (size_t)KVD * DMODEL;
  bf16_t* Wo_b = p; p += (size_t)DMODEL * DMODEL;
  bf16_t* qbuf = p; p += (size_t)SEQ * DMODEL;
  bf16_t* kbuf = p; p += (size_t)SEQ * KVD;
  bf16_t* vbuf = p; p += (size_t)SEQ * KVD;
  bf16_t* vtb  = p; p += (size_t)KVD * SEQ;
  bf16_t* aob  = p;

  // 1) fp32 -> bf16 conversions
  cvt_f32_bf16<<<(SEQ * DMODEL) / 256, 256, 0, stream>>>(xq, xq_b, SEQ * DMODEL);
  cvt_f32_bf16<<<(SEQ * DMODEL) / 256, 256, 0, stream>>>(xk, xk_b, SEQ * DMODEL);
  cvt_f32_bf16<<<(SEQ * DMODEL) / 256, 256, 0, stream>>>(xv, xv_b, SEQ * DMODEL);
  cvt_f32_bf16<<<(DMODEL * DMODEL) / 256, 256, 0, stream>>>(Wq, Wq_b, DMODEL * DMODEL);
  cvt_f32_bf16<<<(KVD * DMODEL) / 256, 256, 0, stream>>>(Wk, Wk_b, KVD * DMODEL);
  cvt_f32_bf16<<<(KVD * DMODEL) / 256, 256, 0, stream>>>(Wv, Wv_b, KVD * DMODEL);
  cvt_f32_bf16<<<(DMODEL * DMODEL) / 256, 256, 0, stream>>>(Wo, Wo_b, DMODEL * DMODEL);

  // 2) projections (x @ W^T) via WMMA + async-LDS weight staging
  gemm_nt<bf16_t><<<dim3(SEQ / 128, DMODEL / 128), 128, 0, stream>>>(
      xq_b, Wq_b, qbuf, SEQ, DMODEL, DMODEL);
  gemm_nt<bf16_t><<<dim3(SEQ / 128, KVD / 128), 128, 0, stream>>>(
      xk_b, Wk_b, kbuf, SEQ, KVD, DMODEL);
  gemm_nt<bf16_t><<<dim3(SEQ / 128, KVD / 128), 128, 0, stream>>>(
      xv_b, Wv_b, vbuf, SEQ, KVD, DMODEL);

  // 3) QK RMSNorm + RoPE (in place)
  rmsnorm_rope<<<(SEQ * NHEAD * 32) / 256, 256, 0, stream>>>(qbuf, qw, NHEAD);
  rmsnorm_rope<<<(SEQ * NKV * 32) / 256, 256, 0, stream>>>(kbuf, kw, NKV);

  // 4) V transpose for contiguous P*V B-fragments
  vtranspose<<<(SEQ * KVD) / 256, 256, 0, stream>>>(vbuf, vtb);

  // 5) sliding-window flash attention
  flash_swa<<<(NHEAD * (SEQ / 16)) / 8, 256, 0, stream>>>(qbuf, kbuf, vtb, aob);

  // 6) output projection -> fp32 d_out
  gemm_nt<float><<<dim3(SEQ / 128, DMODEL / 128), 128, 0, stream>>>(
      aob, Wo_b, (float*)d_out, SEQ, DMODEL, DMODEL);
}
